// SubtleGeneratorUNet_10952166605356
// MI455X (gfx1250) — compile-verified, matrix operand order checked by reading, not run on hardware
//
#include <hip/hip_runtime.h>
#include <hip/hip_bf16.h>

typedef unsigned short u16;
typedef __attribute__((ext_vector_type(16))) __bf16 v16bf;
typedef __attribute__((ext_vector_type(8)))  float  v8f;
typedef __attribute__((ext_vector_type(4)))  unsigned int v4u;
typedef __attribute__((ext_vector_type(8)))  int v8i;
typedef __attribute__((ext_vector_type(4)))  int v4i;

#define DIM    256
#define ROWS   4096     // B*NP tokens
#define MLPD   1024
#define NHEAD  8
#define DHEAD  32
#define NLAYER 2
#define NWIN   49

#define HAS_TDM __has_builtin(__builtin_amdgcn_tensor_load_to_lds)

__device__ __forceinline__ u16 f2bf(float x) {
  unsigned u = __builtin_bit_cast(unsigned, x);
  u += 0x7fffu + ((u >> 16) & 1u);          // round-to-nearest-even
  return (u16)(u >> 16);
}
__device__ __forceinline__ float bf2f(u16 x) {
  return __builtin_bit_cast(float, ((unsigned)x) << 16);
}
__device__ __forceinline__ float gelu_tanh(float x) {
  float t = tanhf(0.7978845608028654f * (x + 0.044715f * x * x * x));
  return 0.5f * x * (1.0f + t);
}

// ---------------------------------------------------------------------------
// Patch embed: grid[n,d] = sum_{p,q} x[i*4+p, j*4+q] * w[d,p,q] + b[d]
// ---------------------------------------------------------------------------
__global__ __launch_bounds__(256) void patch_embed_k(
    const float* __restrict__ x, const float* __restrict__ w,
    const float* __restrict__ b, float* __restrict__ grid,
    u16* __restrict__ grid_bf, float* __restrict__ h) {
  int idx = blockIdx.x * 256 + threadIdx.x;   // n*256 + d
  int d = idx & 255;
  int n = idx >> 8;
  int i = n >> 6, j = n & 63;
  float acc = b[d];
  const float* wp = w + d * 16;
#pragma unroll
  for (int p = 0; p < 4; ++p)
#pragma unroll
    for (int q = 0; q < 4; ++q)
      acc += x[(i * 4 + p) * 256 + (j * 4 + q)] * wp[p * 4 + q];
  grid[idx] = acc;
  h[idx] = acc;
  grid_bf[idx] = f2bf(acc);
}

// ---------------------------------------------------------------------------
// fp32 [K][N] weight -> bf16 transposed [N][K]
// ---------------------------------------------------------------------------
__global__ __launch_bounds__(256) void xpose_bf16_k(
    const float* __restrict__ src, u16* __restrict__ dst, int K, int N) {
  int idx = blockIdx.x * 256 + threadIdx.x;
  if (idx >= K * N) return;
  int k = idx / N, n = idx - k * N;
  dst[(size_t)n * K + k] = f2bf(src[idx]);
}

// ---------------------------------------------------------------------------
// LayerNorm over D=256, one row per 256-thread block, bf16 output.
// ---------------------------------------------------------------------------
__global__ __launch_bounds__(256) void layernorm_bf16_k(
    const float* __restrict__ xin, const float* __restrict__ g,
    const float* __restrict__ b, u16* __restrict__ outp) {
  __shared__ float red[8];
  int row = blockIdx.x, tid = threadIdx.x;
  float x = xin[(size_t)row * 256 + tid];
  float s = x;
  for (int o = 16; o; o >>= 1) s += __shfl_xor(s, o, 32);
  if ((tid & 31) == 0) red[tid >> 5] = s;
  __syncthreads();
  float mean = (red[0] + red[1] + red[2] + red[3] +
                red[4] + red[5] + red[6] + red[7]) * (1.0f / 256.0f);
  __syncthreads();
  float d = x - mean;
  float v = d * d;
  for (int o = 16; o; o >>= 1) v += __shfl_xor(v, o, 32);
  if ((tid & 31) == 0) red[tid >> 5] = v;
  __syncthreads();
  float var = (red[0] + red[1] + red[2] + red[3] +
               red[4] + red[5] + red[6] + red[7]) * (1.0f / 256.0f);
  float r = rsqrtf(var + 1e-6f);
  outp[(size_t)row * 256 + tid] = f2bf(d * r * g[tid] + b[tid]);
}

// ---------------------------------------------------------------------------
// TDM: DMA one 64-row x 32-bf16 tile (64B rows) into LDS with a 16B pad
// every 64B so the LDS pitch is 40 u16 (bank-spread for fragment reads).
// data_size=8B, tile_dim0=8, tile_dim1=64, stride = K/4 (8B units).
// ---------------------------------------------------------------------------
#if HAS_TDM
__device__ __forceinline__ void tdm_load_tile_64x32(
    unsigned lds_addr, const u16* gsrc, int pitch_elems /*K in bf16*/) {
  unsigned long long ga = (unsigned long long)(size_t)gsrc;
  v4u g0;
  g0[0] = 1u;                                   // count=1 valid descriptor
  g0[1] = lds_addr;                             // LDS byte address
  g0[2] = (unsigned)ga;                         // global_addr[31:0]
  g0[3] = (unsigned)(ga >> 32) | (2u << 30);    // global_addr[56:32] | type=2
  unsigned stride8 = (unsigned)(pitch_elems >> 2);   // row pitch in 8B units
  v8i g1;
  // data_size=3(8B) | pad_enable | pad_interval=3(16DW) | pad_amount=3(4DW)
  g1[0] = (3 << 16) | (1 << 20) | (3 << 22) | (3 << 25);
  g1[1] = (int)(0x7fffu << 16);                 // tensor_dim0[15:0]=0x7fff
  g1[2] = (int)(0x7fffu | (0x7fffu << 16));     // td0[31:16] | td1[15:0]
  g1[3] = (int)(0x7fffu | (8u << 16));          // td1[31:16] | tile_dim0=8
  g1[4] = 64;                                   // tile_dim1=64, tile_dim2=0
  g1[5] = (int)stride8;                         // tensor_dim0_stride[31:0]
  g1[6] = 0;                                    // stride[47:32]=0, dim1_stride lo=0
  g1[7] = 0;
  v4i z4 = {};
#if __clang_major__ >= 23
  v8i z8 = {};
  __builtin_amdgcn_tensor_load_to_lds(g0, g1, z4, z4, z8, 0);
#else
  __builtin_amdgcn_tensor_load_to_lds(g0, g1, z4, z4, 0);
#endif
}
#endif

// ---------------------------------------------------------------------------
// Shared WMMA micro-step: 2x2 16x16 tiles from a 64x(32 @pitch 40) LDS pair.
// ---------------------------------------------------------------------------
union Frag32 { uint4 u[2]; v16bf v; };

__device__ __forceinline__ void wmma_step(
    const u16* __restrict__ Asb, const u16* __restrict__ Bsb,
    int wr, int wc, int l16, int hf, v8f acc[2][2]) {
  v16bf af[2], bfr[2];
#pragma unroll
  for (int i = 0; i < 2; ++i) {
    // A frag: lane holds row M=l16 (+16/subtile); elems 0..7 -> K=hf*8+e,
    // elems 8..15 -> K=16+hf*8+(e-8). B mirrors A (B is stored [N][K]).
    const u16* pa = &Asb[(wr + i * 16 + l16) * 40];
    Frag32 fa;
    fa.u[0] = *(const uint4*)(pa + hf * 8);
    fa.u[1] = *(const uint4*)(pa + 16 + hf * 8);
    af[i] = fa.v;
    const u16* pb = &Bsb[(wc + i * 16 + l16) * 40];
    Frag32 fb;
    fb.u[0] = *(const uint4*)(pb + hf * 8);
    fb.u[1] = *(const uint4*)(pb + 16 + hf * 8);
    bfr[i] = fb.v;
  }
#pragma unroll
  for (int i = 0; i < 2; ++i)
#pragma unroll
    for (int j = 0; j < 2; ++j)
      acc[i][j] = __builtin_amdgcn_wmma_f32_16x16x32_bf16(
          false, af[i], false, bfr[j], (short)0, acc[i][j], false, false);
}

// ---------------------------------------------------------------------------
// bf16 WMMA GEMM: out[M][N] = A[M][K] @ Bt[N][K]^T + bias (+ epilogue)
// 128 threads = 4 waves; block tile 64x64; wave tile 32x32 (2x2 WMMA).
// TDM path: wave0 DMAs double-buffered 64x32 tiles, tracked by TENSORcnt.
// mode 0: f32 out  1: bf16 out  2: bf16 gelu  3: f32 resid + acc + bias
// ---------------------------------------------------------------------------
__global__ __launch_bounds__(128) void gemm_bf16_wmma_k(
    const u16* __restrict__ A, const u16* __restrict__ Bt,
    const float* __restrict__ bias, void* __restrict__ outp,
    const float* __restrict__ resid, int Mdim, int Ndim, int Kdim, int mode) {
  __shared__ __align__(16) u16 sm[4][64 * 40];   // A0 A1 B0 B1
  const int tid  = threadIdx.x;
  const int wave = tid >> 5, lane = tid & 31;
  const int m0 = blockIdx.y << 6, n0 = blockIdx.x << 6;
  const int wr = (wave >> 1) << 5, wc = (wave & 1) << 5;
  const int hf = lane >> 4, l16 = lane & 15;
  v8f acc[2][2] = {};
#if HAS_TDM
  unsigned aLds[2] = { (unsigned)(size_t)&sm[0][0], (unsigned)(size_t)&sm[1][0] };
  unsigned bLds[2] = { (unsigned)(size_t)&sm[2][0], (unsigned)(size_t)&sm[3][0] };
  if (tid < 32) {   // wave 0 issues DMA (TDM ignores EXEC; one issue per WG)
    tdm_load_tile_64x32(aLds[0], A + (size_t)m0 * Kdim, Kdim);
    tdm_load_tile_64x32(bLds[0], Bt + (size_t)n0 * Kdim, Kdim);
  }
  const int steps = Kdim >> 5;
  for (int s = 0; s < steps; ++s) {
    const int cur = s & 1;
    if (s + 1 < steps) {
      if (tid < 32) {
        const int kt = (s + 1) << 5;
        tdm_load_tile_64x32(aLds[cur ^ 1], A + (size_t)m0 * Kdim + kt, Kdim);
        tdm_load_tile_64x32(bLds[cur ^ 1], Bt + (size_t)n0 * Kdim + kt, Kdim);
      }
      // 2 newer loads in flight; in-order completion => current tile ready
      __builtin_amdgcn_s_wait_tensorcnt(2);
    } else {
      __builtin_amdgcn_s_wait_tensorcnt(0);
    }
    __syncthreads();                       // tile visible to all waves
    wmma_step(&sm[cur][0], &sm[2 + cur][0], wr, wc, l16, hf, acc);
    __syncthreads();                       // done reading before DMA reuse
  }
#else
  const int rA = tid >> 1, cA = (tid & 1) << 4;  // 64 rows x 2 chunks of 16
  for (int kt = 0; kt < Kdim; kt += 32) {
    const uint4* gA = (const uint4*)(A + (size_t)(m0 + rA) * Kdim + kt + cA);
    *(uint4*)&sm[0][rA * 40 + cA]     = gA[0];
    *(uint4*)&sm[0][rA * 40 + cA + 8] = gA[1];
    const uint4* gB = (const uint4*)(Bt + (size_t)(n0 + rA) * Kdim + kt + cA);
    *(uint4*)&sm[2][rA * 40 + cA]     = gB[0];
    *(uint4*)&sm[2][rA * 40 + cA + 8] = gB[1];
    __syncthreads();
    wmma_step(&sm[0][0], &sm[2][0], wr, wc, l16, hf, acc);
    __syncthreads();
  }
#endif
  // Epilogue: C layout: VGPR r -> row (hf*8 + r), col = l16
#pragma unroll
  for (int i = 0; i < 2; ++i)
#pragma unroll
    for (int j = 0; j < 2; ++j) {
      int col = n0 + wc + j * 16 + l16;
      float bv = bias[col];
#pragma unroll
      for (int r = 0; r < 8; ++r) {
        int row = m0 + wr + i * 16 + hf * 8 + r;
        size_t idx = (size_t)row * Ndim + col;
        float v = acc[i][j][r] + bv;
        if (mode == 0)      ((float*)outp)[idx] = v;
        else if (mode == 1) ((u16*)outp)[idx] = f2bf(v);
        else if (mode == 2) ((u16*)outp)[idx] = f2bf(gelu_tanh(v));
        else                ((float*)outp)[idx] = resid[idx] + v;
      }
    }
}

// ---------------------------------------------------------------------------
// Cross attention over the 7x7 window, one wave per (token, head).
// K/V are pre-projected per grid row; padded rows use the projection bias
// (zero-padded input row projects to exactly the bias vector).
// ---------------------------------------------------------------------------
__global__ __launch_bounds__(256) void cross_attn_k(
    const float* __restrict__ q, const float* __restrict__ Kg,
    const float* __restrict__ Vg, const float* __restrict__ bk,
    const float* __restrict__ bv, const float* __restrict__ rbias,
    u16* __restrict__ ctx_bf) {
  int n = blockIdx.x;
  int head = threadIdx.x >> 5, lane = threadIdx.x & 31;
  int i = n >> 6, j = n & 63;
  const float scale = 0.17677669529663687f;     // 1/sqrt(32)
  float qv = q[(size_t)n * 256 + head * 32 + lane];
  float s0 = -3.0e38f, s1 = -3.0e38f;
  for (int w = 0; w < NWIN; ++w) {
    int wi = w / 7, wj = w - wi * 7;
    int gi = i + wi - 3, gj = j + wj - 3;
    bool inb = (gi >= 0) & (gi < 64) & (gj >= 0) & (gj < 64);
    const float* kr = inb ? (Kg + ((size_t)(gi * 64 + gj)) * 256 + head * 32)
                          : (bk + head * 32);
    float p = qv * kr[lane];
    for (int o = 16; o; o >>= 1) p += __shfl_xor(p, o, 32);
    float sc = p * scale + rbias[head * NWIN + w];
    if (w < 32) { if (lane == w) s0 = sc; }
    else        { if (lane == w - 32) s1 = sc; }
  }
  float m = fmaxf(s0, s1);
  for (int o = 16; o; o >>= 1) m = fmaxf(m, __shfl_xor(m, o, 32));
  float e0 = __expf(s0 - m);
  float e1 = (lane < NWIN - 32) ? __expf(s1 - m) : 0.0f;
  float sum = e0 + e1;
  for (int o = 16; o; o >>= 1) sum += __shfl_xor(sum, o, 32);
  float accv = 0.0f;
  for (int w = 0; w < NWIN; ++w) {
    int wi = w / 7, wj = w - wi * 7;
    int gi = i + wi - 3, gj = j + wj - 3;
    bool inb = (gi >= 0) & (gi < 64) & (gj >= 0) & (gj < 64);
    const float* vr = inb ? (Vg + ((size_t)(gi * 64 + gj)) * 256 + head * 32)
                          : (bv + head * 32);
    float pw = (w < 32) ? __shfl(e0, w, 32) : __shfl(e1, w - 32, 32);
    accv += pw * vr[lane];
  }
  ctx_bf[(size_t)n * 256 + head * 32 + lane] = f2bf(accv / sum);
}

// ---------------------------------------------------------------------------
// fc head (N=16) fused with the inverse patchify permutation.
// y[i*4+p][j*4+q] = lat[n=(i,j)][c=(p,q)]
// ---------------------------------------------------------------------------
__global__ __launch_bounds__(256) void fc_out_k(
    const u16* __restrict__ lnf, const float* __restrict__ fcw,
    const float* __restrict__ fcb, float* __restrict__ outp) {
  int idx = blockIdx.x * 256 + threadIdx.x;    // n*16 + c
  int c = idx & 15, n = idx >> 4;
  float acc = fcb[c];
  const u16* row = lnf + (size_t)n * 256;
  for (int d = 0; d < 256; ++d) acc += bf2f(row[d]) * fcw[d * 16 + c];
  int i = n >> 6, j = n & 63, p = c >> 2, qq = c & 3;
  outp[(size_t)(i * 4 + p) * 256 + (j * 4 + qq)] = acc;
}

// ---------------------------------------------------------------------------
extern "C" void kernel_launch(void* const* d_in, const int* in_sizes, int n_in,
                              void* d_out, int out_size, void* d_ws, size_t ws_size,
                              hipStream_t stream) {
  (void)in_sizes; (void)n_in; (void)out_size; (void)ws_size;
  const float* x        = (const float*)d_in[0];
  const float* conv_w   = (const float*)d_in[1];
  const float* conv_b   = (const float*)d_in[2];
  const float* slf_wqkv = (const float*)d_in[3];
  const float* slf_bqkv = (const float*)d_in[4];
  const float* slf_wo   = (const float*)d_in[5];
  const float* slf_bo   = (const float*)d_in[6];
  const float* enc_wqkv = (const float*)d_in[7];
  const float* enc_bqkv = (const float*)d_in[8];
  const float* enc_wo   = (const float*)d_in[9];
  const float* enc_bo   = (const float*)d_in[10];
  const float* rel_bias = (const float*)d_in[11];
  const float* ln_g     = (const float*)d_in[12];
  const float* ln_b     = (const float*)d_in[13];
  const float* mlp_w1   = (const float*)d_in[14];
  const float* mlp_b1   = (const float*)d_in[15];
  const float* mlp_w2   = (const float*)d_in[16];
  const float* mlp_b2   = (const float*)d_in[17];
  const float* dn_g     = (const float*)d_in[18];
  const float* dn_b     = (const float*)d_in[19];
  const float* fc_w     = (const float*)d_in[20];
  const float* fc_b     = (const float*)d_in[21];
  float* out = (float*)d_out;

  // ---- workspace carve ----
  size_t off = 0;
  auto carve = [&](size_t bytes) {
    void* p = (char*)d_ws + off;
    off += (bytes + 255) & ~(size_t)255;
    return p;
  };
  float* gridF  = (float*)carve((size_t)ROWS * DIM * 4);
  u16*   gridBf = (u16*)  carve((size_t)ROWS * DIM * 2);
  float* hbuf   = (float*)carve((size_t)ROWS * DIM * 4);
  u16*   lnBf   = (u16*)  carve((size_t)ROWS * DIM * 2);
  float* qB     = (float*)carve((size_t)ROWS * DIM * 4);
  float* KgB    = (float*)carve((size_t)ROWS * DIM * 4);
  float* VgB    = (float*)carve((size_t)ROWS * DIM * 4);
  u16*   abuf   = (u16*)  carve((size_t)ROWS * DIM * 2);   // v_self / ctx
  u16*   hidBf  = (u16*)  carve((size_t)ROWS * MLPD * 2);
  const size_t PER_LAYER = 6 * 65536 + 2 * 262144;
  u16*   wPool  = (u16*)  carve((size_t)NLAYER * PER_LAYER * 2);

  auto xpose = [&](const float* src, u16* dst, int K, int N) {
    int total = K * N;
    xpose_bf16_k<<<(total + 255) / 256, 256, 0, stream>>>(src, dst, K, N);
  };
  auto gemm = [&](const u16* A, const u16* Bt, const float* bias, void* op,
                  const float* resid, int Md, int Nd, int Kd, int mode) {
    dim3 g(Nd / 64, Md / 64);
    gemm_bf16_wmma_k<<<g, 128, 0, stream>>>(A, Bt, bias, op, resid, Md, Nd, Kd, mode);
  };
  auto lnorm = [&](const float* xin, const float* g, const float* b, u16* op) {
    layernorm_bf16_k<<<ROWS, 256, 0, stream>>>(xin, g, b, op);
  };

  // ---- patch embed (also inits residual stream) ----
  patch_embed_k<<<ROWS, 256, 0, stream>>>(x, conv_w, conv_b, gridF, gridBf, hbuf);

  for (int l = 0; l < NLAYER; ++l) {
    u16* wv_s = wPool + (size_t)l * PER_LAYER;
    u16* wo_s = wv_s + 65536;
    u16* wq_e = wo_s + 65536;
    u16* wk_e = wq_e + 65536;
    u16* wv_e = wk_e + 65536;
    u16* wo_e = wv_e + 65536;
    u16* w1t  = wo_e + 65536;
    u16* w2t  = w1t + 262144;
    // weight prep (bf16, transposed [N][K])
    xpose(slf_wqkv + ((size_t)(l * 3 + 2)) * 65536, wv_s, 256, 256);
    xpose(slf_wo   + (size_t)l * 65536,             wo_s, 256, 256);
    xpose(enc_wqkv + ((size_t)(l * 3 + 0)) * 65536, wq_e, 256, 256);
    xpose(enc_wqkv + ((size_t)(l * 3 + 1)) * 65536, wk_e, 256, 256);
    xpose(enc_wqkv + ((size_t)(l * 3 + 2)) * 65536, wv_e, 256, 256);
    xpose(enc_wo   + (size_t)l * 65536,             wo_e, 256, 256);
    xpose(mlp_w1   + (size_t)l * 262144,            w1t,  256, 1024);
    xpose(mlp_w2   + (size_t)l * 262144,            w2t, 1024, 256);

    // --- self attention (seq len 1 -> softmax==1 -> ctx = v) ---
    lnorm(hbuf, ln_g + (size_t)(l * 3 + 0) * 256, ln_b + (size_t)(l * 3 + 0) * 256, lnBf);
    gemm(lnBf, wv_s, slf_bqkv + (size_t)(l * 3 + 2) * 256, abuf, nullptr,
         ROWS, 256, 256, 1);
    gemm(abuf, wo_s, slf_bo + (size_t)l * 256, hbuf, hbuf, ROWS, 256, 256, 3);

    // --- cross attention over 7x7 window (K/V projected on grid rows) ---
    lnorm(hbuf, ln_g + (size_t)(l * 3 + 1) * 256, ln_b + (size_t)(l * 3 + 1) * 256, lnBf);
    gemm(lnBf,   wq_e, enc_bqkv + (size_t)(l * 3 + 0) * 256, qB,  nullptr, ROWS, 256, 256, 0);
    gemm(gridBf, wk_e, enc_bqkv + (size_t)(l * 3 + 1) * 256, KgB, nullptr, ROWS, 256, 256, 0);
    gemm(gridBf, wv_e, enc_bqkv + (size_t)(l * 3 + 2) * 256, VgB, nullptr, ROWS, 256, 256, 0);
    cross_attn_k<<<ROWS, 256, 0, stream>>>(
        qB, KgB, VgB,
        enc_bqkv + (size_t)(l * 3 + 1) * 256,
        enc_bqkv + (size_t)(l * 3 + 2) * 256,
        rel_bias + (size_t)l * NHEAD * NWIN, abuf);
    gemm(abuf, wo_e, enc_bo + (size_t)l * 256, hbuf, hbuf, ROWS, 256, 256, 3);

    // --- MLP ---
    lnorm(hbuf, ln_g + (size_t)(l * 3 + 2) * 256, ln_b + (size_t)(l * 3 + 2) * 256, lnBf);
    gemm(lnBf, w1t, mlp_b1 + (size_t)l * 1024, hidBf, nullptr, ROWS, 1024, 256, 2);
    gemm(hidBf, w2t, mlp_b2 + (size_t)l * 256, hbuf, hbuf, ROWS, 256, 1024, 3);
  }

  // ---- head ----
  lnorm(hbuf, dn_g, dn_b, lnBf);
  fc_out_k<<<ROWS * 16 / 256, 256, 0, stream>>>(lnBf, fc_w, fc_b, out);
}
